// MoELayer_6485400616970
// MI455X (gfx1250) — matrix operand order repaired and checked by reading, not compiled, not run on hardware
//
#include <hip/hip_runtime.h>
#include <hip/hip_bf16.h>
#include <math.h>

// ---------------- problem constants ----------------
constexpr int T  = 4096;   // tokens (B*S)
constexpr int C  = 1024;   // model dim
constexpr int F  = 4096;   // ffn dim
constexpr int E  = 8;      // experts
// tiling
constexpr int BM  = 64;    // tokens per workgroup tile (4 m-tiles of 16)
constexpr int FC  = 256;   // F-chunk (16 waves x 16 cols)
constexpr int NW  = 16;    // waves per workgroup (512 threads)
constexpr int XLD = C + 8;   // bf16 row stride of x tile (rows 16B aligned)
constexpr int HLD = FC + 8;  // bf16 row stride of h chunk
constexpr int YLD = C + 4;   // f32 row stride of y staging

constexpr int SX_BYTES = BM * XLD * 2;                 // 132096
constexpr int SH_BYTES = BM * HLD * 2;                 // 33792
constexpr int SY_BYTES = BM * YLD * 4;                 // 263168
constexpr int SMEM_BYTES = (SX_BYTES + SH_BYTES) > SY_BYTES
                               ? (SX_BYTES + SH_BYTES) : SY_BYTES;  // 263168

typedef __attribute__((ext_vector_type(16))) __bf16 v16bf;
typedef __attribute__((ext_vector_type(8)))  __bf16 v8bf;
typedef __attribute__((ext_vector_type(8)))  float  v8f;

union BF16x16 { v16bf v; v8bf h[2]; };

// A-fragment (16x32 bf16): lane L holds row m=L%16, k in [kh*16,kh*16+16)
// where kh=L/16.  p points at row_base + k + kh*16.
__device__ inline v16bf load_a_frag(const __bf16* p) {
  BF16x16 u;
  u.h[0] = *(const v8bf*)(p);
  u.h[1] = *(const v8bf*)(p + 8);
  return u.v;
}

// B-fragment (32x16 bf16), bf16 weights: lane L holds col n=L%16,
// k=(L/16)*16+0..15 packed in memory order -> 2x 16B loads.
__device__ inline v16bf load_b_frag_bf16(const __bf16* __restrict__ p) {
  BF16x16 u;
  u.h[0] = *(const v8bf*)(p);
  u.h[1] = *(const v8bf*)(p + 8);
  return u.v;
}

// Fallback: convert 16 consecutive f32 of a weight row to bf16 on the fly.
__device__ inline v16bf load_b_frag_f32(const float* __restrict__ p) {
  v16bf r;
#pragma unroll
  for (int i = 0; i < 16; i += 4) {
    float4 f = *(const float4*)(p + i);
    r[i + 0] = (__bf16)f.x;
    r[i + 1] = (__bf16)f.y;
    r[i + 2] = (__bf16)f.z;
    r[i + 3] = (__bf16)f.w;
  }
  return r;
}

__device__ inline v8f wmma_bf16(v16bf a, v16bf b, v8f c) {
  return __builtin_amdgcn_wmma_f32_16x16x32_bf16(false, a, false, b, (short)0, c,
                                                 false, false);
}

__device__ inline float gelu_exact(float v) {
  return 0.5f * v * (1.0f + erff(v * 0.70710678118654752440f));
}

// CDNA5 async global->LDS copy, 16B per lane, tracked by ASYNCcnt.
__device__ inline void async_g2l_b128(unsigned lds_off, const void* gptr) {
  asm volatile("global_load_async_to_lds_b128 %0, %1, off"
               :: "v"(lds_off), "v"(gptr)
               : "memory");
}
__device__ inline void wait_asynccnt0() {
  asm volatile("s_wait_asynccnt 0x0" ::: "memory");
}

// ---------------- fp32 -> bf16 bulk conversion ----------------
__global__ __launch_bounds__(256) void cvt_f32_to_bf16(
    const float* __restrict__ src, __bf16* __restrict__ dst, int n) {
  int i = (blockIdx.x * 256 + threadIdx.x) * 8;
  if (i < n) {
    float4 a = *(const float4*)(src + i);
    float4 b = *(const float4*)(src + i + 4);
    v8bf o;
    o[0] = (__bf16)a.x; o[1] = (__bf16)a.y; o[2] = (__bf16)a.z; o[3] = (__bf16)a.w;
    o[4] = (__bf16)b.x; o[5] = (__bf16)b.y; o[6] = (__bf16)b.z; o[7] = (__bf16)b.w;
    *(v8bf*)(dst + i) = o;
  }
}

// ---------------- router: logits/softmax/top2/comb/lists ----------------
__global__ __launch_bounds__(256) void moe_router(
    const float* __restrict__ x, const float* __restrict__ gw,
    float* __restrict__ comb, float* __restrict__ probs,
    int* __restrict__ counts, int* __restrict__ lists) {
  __shared__ float sGW[E * C];  // 32 KB
  const int tid = threadIdx.x;
  for (int i = tid; i < E * C; i += 256) sGW[i] = gw[i];
  __syncthreads();

  const int wave = tid >> 5, lane = tid & 31;
  const int t = blockIdx.x * 8 + wave;
  const float* xr = x + (size_t)t * C;

  float acc[E];
#pragma unroll
  for (int e = 0; e < E; ++e) acc[e] = 0.0f;
  for (int c = lane; c < C; c += 32) {
    float xv = xr[c];
#pragma unroll
    for (int e = 0; e < E; ++e) acc[e] += xv * sGW[e * C + c];
  }
#pragma unroll
  for (int e = 0; e < E; ++e) {
#pragma unroll
    for (int off = 16; off > 0; off >>= 1) acc[e] += __shfl_xor(acc[e], off, 32);
  }

  if (lane == 0) {
    float mx = acc[0];
#pragma unroll
    for (int e = 1; e < E; ++e) mx = fmaxf(mx, acc[e]);
    float p[E], s = 0.0f;
#pragma unroll
    for (int e = 0; e < E; ++e) { p[e] = expf(acc[e] - mx); s += p[e]; }
    float inv = 1.0f / s;
#pragma unroll
    for (int e = 0; e < E; ++e) { p[e] *= inv; probs[(size_t)t * E + e] = p[e]; }
    // top-2, lowest index wins ties (matches jax.lax.top_k)
    int e0 = 0;
#pragma unroll
    for (int e = 1; e < E; ++e) if (p[e] > p[e0]) e0 = e;
    int e1 = (e0 == 0) ? 1 : 0;
#pragma unroll
    for (int e = 0; e < E; ++e) if (e != e0 && p[e] > p[e1]) e1 = e;
    float g0 = p[e0], g1 = p[e1];
    float d = g0 + g1 + 1e-9f;
    g0 /= d; g1 /= d;
#pragma unroll
    for (int e = 0; e < E; ++e)
      comb[(size_t)t * E + e] = (e == e0) ? g0 : ((e == e1) ? g1 : 0.0f);
    int p0 = atomicAdd(&counts[e0 * 2 + 0], 1);
    lists[(size_t)(e0 * 2 + 0) * T + p0] = t;
    int p1 = atomicAdd(&counts[e1 * 2 + 1], 1);
    lists[(size_t)(e1 * 2 + 1) * T + p1] = t;
  }
}

// ---------------- deterministic probs column sums ----------------
__global__ __launch_bounds__(256) void moe_probs_sum(const float* __restrict__ probs,
                                                     float* __restrict__ psum) {
  __shared__ float red[256];
  const int e = blockIdx.x, tid = threadIdx.x;
  float s = 0.0f;
  for (int t = tid; t < T; t += 256) s += probs[(size_t)t * E + e];
  red[tid] = s;
  __syncthreads();
  for (int w = 128; w > 0; w >>= 1) {
    if (tid < w) red[tid] += red[tid + w];
    __syncthreads();
  }
  if (tid == 0) psum[e] = red[0];
}

__global__ void moe_aux(const int* __restrict__ counts, const float* __restrict__ psum,
                        float* __restrict__ out_aux) {
  if (threadIdx.x == 0 && blockIdx.x == 0) {
    float a = 0.0f;
    for (int e = 0; e < E; ++e) {
      float f = (float)(counts[e * 2] + counts[e * 2 + 1]) / ((float)(T * 2) + 1e-9f);
      float p = psum[e] / (float)T;
      a += f * p;
    }
    *out_aux = (float)E * a;
  }
}

// ---------------- fused expert GEMM (WMMA bf16) ----------------
// Grid: (T/BM, E). 512 threads = 16 waves. Wave w owns columns
// [w*64, w*64+64) of C (4 n-tiles) x 4 m-tiles = 16 persistent f32 accs.
// slot 0: out = comb*(y+pb)   (each token exactly once -> full overwrite)
// slot 1: out += comb*(y+pb)  (each token exactly once -> race-free RMW)
template <bool PRECONV>
__global__ __launch_bounds__(512) void moe_expert_pass(
    const float* __restrict__ x,   const __bf16* __restrict__ xb,
    const float* __restrict__ fcw, const __bf16* __restrict__ fcwb,
    const float* __restrict__ fcb,
    const float* __restrict__ pw,  const __bf16* __restrict__ pwb,
    const float* __restrict__ pb,  const float* __restrict__ comb,
    const int* __restrict__ counts, const int* __restrict__ lists,
    float* __restrict__ out, int slot) {
  extern __shared__ char smem[];
  __bf16* sX = (__bf16*)smem;                    // BM * XLD bf16
  __bf16* sH = (__bf16*)(smem + SX_BYTES);       // BM * HLD bf16
  float*  sY = (float*)smem;                     // BM * YLD f32 (aliases sX/sH)
  __shared__ float sComb[BM];
  __shared__ int   sTok[BM];

  const int e    = blockIdx.y;
  const int list = e * 2 + slot;
  const int n    = counts[list];
  const int tile = blockIdx.x;
  if (tile * BM >= n) return;

  const int tid  = threadIdx.x;
  const int lane = tid & 31;
  const int wave = tid >> 5;

  if (tid < BM) {
    int idx = tile * BM + tid;
    int tok = (idx < n) ? lists[(size_t)list * T + idx] : -1;
    sTok[tid]  = tok;
    sComb[tid] = (tok >= 0) ? comb[(size_t)tok * E + e] : 0.0f;
  }
  __syncthreads();

  // stage x tile (gathered rows) -> bf16 LDS; zero-pad invalid rows
  {
    const int r   = tid >> 3;              // 64 rows, 8 threads/row
    const int c0  = (tid & 7) * (C / 8);   // 128 cols each
    const int tok = sTok[r];
    if (PRECONV) {
      if (tok >= 0) {
        const __bf16* src = xb + (size_t)tok * C + c0;
#pragma unroll
        for (int j = 0; j < C / 8; j += 8) {  // 16 x 16B async copies
          unsigned lds_off = (unsigned)(((size_t)r * XLD + c0 + j) * sizeof(__bf16));
          async_g2l_b128(lds_off, (const void*)(src + j));
        }
      } else {
        v8bf zv;
#pragma unroll
        for (int q = 0; q < 8; ++q) zv[q] = (__bf16)0.0f;
        for (int j = 0; j < C / 8; j += 8)
          *(v8bf*)&sX[(size_t)r * XLD + c0 + j] = zv;
      }
      wait_asynccnt0();
    } else {
      const float* src = x + (size_t)(tok < 0 ? 0 : tok) * C + c0;
      for (int j = 0; j < C / 8; j += 8) {
        v8bf hv;
        if (tok >= 0) {
          float4 f0 = *(const float4*)(src + j);
          float4 f1 = *(const float4*)(src + j + 4);
          hv[0] = (__bf16)f0.x; hv[1] = (__bf16)f0.y;
          hv[2] = (__bf16)f0.z; hv[3] = (__bf16)f0.w;
          hv[4] = (__bf16)f1.x; hv[5] = (__bf16)f1.y;
          hv[6] = (__bf16)f1.z; hv[7] = (__bf16)f1.w;
        } else {
#pragma unroll
          for (int q = 0; q < 8; ++q) hv[q] = (__bf16)0.0f;
        }
        *(v8bf*)&sX[(size_t)r * XLD + c0 + j] = hv;
      }
    }
  }
  __syncthreads();

  v8f yacc[16];  // [mt*4 + nt]
  const v8f vzero = {0.f, 0.f, 0.f, 0.f, 0.f, 0.f, 0.f, 0.f};
#pragma unroll
  for (int i = 0; i < 16; ++i) yacc[i] = vzero;

  const int bN    = lane & 15;          // B col / C-D col / A row (within 16)
  const int bK    = (lane >> 4) * 16;   // B k base
  const int halfk = (lane >> 4) * 16;   // A k sub-offset

  const float*  fcw_e  = fcw  ? fcw  + (size_t)e * F * C : nullptr;
  const float*  pw_e   = pw   ? pw   + (size_t)e * C * F : nullptr;
  const __bf16* fcwb_e = fcwb ? fcwb + (size_t)e * F * C : nullptr;
  const __bf16* pwb_e  = pwb  ? pwb  + (size_t)e * C * F : nullptr;

  // A rows for the 4 m-tiles (x tile), fixed per lane
  const __bf16* xrow[4];
#pragma unroll
  for (int mt = 0; mt < 4; ++mt) xrow[mt] = sX + (size_t)(mt * 16 + bN) * XLD;

  for (int fc = 0; fc < F; fc += FC) {
    // ---- Phase A: h[:, fc + wave*16 .. +16) = gelu(x @ fcw^T + b) ----
    const int fbase = fc + wave * 16;
    v8f ha[4];
#pragma unroll
    for (int mt = 0; mt < 4; ++mt) ha[mt] = vzero;

    const float*  wrow  = PRECONV ? nullptr : fcw_e + (size_t)(fbase + bN) * C + bK;
    const __bf16* wrowb = PRECONV ? fcwb_e + (size_t)(fbase + bN) * C + bK : nullptr;

    // double-buffered B stream over k
    v16bf bcur = PRECONV ? load_b_frag_bf16(wrowb) : load_b_frag_f32(wrow);
#pragma unroll 2
    for (int k = 0; k < C; k += 32) {
      const int kn = (k + 32 < C) ? (k + 32) : k;
      v16bf bnext = PRECONV ? load_b_frag_bf16(wrowb + kn) : load_b_frag_f32(wrow + kn);
      v16bf a[4];
#pragma unroll
      for (int mt = 0; mt < 4; ++mt) a[mt] = load_a_frag(xrow[mt] + k + halfk);
#pragma unroll
      for (int mt = 0; mt < 4; ++mt) ha[mt] = wmma_bf16(a[mt], bcur, ha[mt]);
      bcur = bnext;
    }
    const float bias = fcb[(size_t)e * F + fbase + bN];
#pragma unroll
    for (int mt = 0; mt < 4; ++mt) {
#pragma unroll
      for (int i = 0; i < 8; ++i) {
        int m = mt * 16 + (lane >> 4) * 8 + i;
        sH[(size_t)m * HLD + wave * 16 + bN] = (__bf16)gelu_exact(ha[mt][i] + bias);
      }
    }
    __syncthreads();

    // ---- Phase B: y[:, wave*64 .. +64) += h_chunk @ pw^T ----
    const int cbase = wave * 64;
#pragma unroll 2
    for (int k = 0; k < FC; k += 32) {
      // A frags (shared across the 4 n-tiles)
      v16bf a[4];
#pragma unroll
      for (int mt = 0; mt < 4; ++mt)
        a[mt] = load_a_frag(sH + (size_t)(mt * 16 + bN) * HLD + k + halfk);
      // batch-issue the 4 B frags, then 16 WMMAs drain with descending waits
      v16bf b[4];
#pragma unroll
      for (int nt = 0; nt < 4; ++nt) {
        if (PRECONV)
          b[nt] = load_b_frag_bf16(pwb_e + (size_t)(cbase + nt * 16 + bN) * F + fc + k + bK);
        else
          b[nt] = load_b_frag_f32(pw_e + (size_t)(cbase + nt * 16 + bN) * F + fc + k + bK);
      }
#pragma unroll
      for (int nt = 0; nt < 4; ++nt)
#pragma unroll
        for (int mt = 0; mt < 4; ++mt)
          yacc[mt * 4 + nt] = wmma_bf16(a[mt], b[nt], yacc[mt * 4 + nt]);
    }
    __syncthreads();  // sH reused next chunk
  }

  // dump accumulators to LDS (sY aliases sX/sH: safe after final barrier)
#pragma unroll
  for (int mt = 0; mt < 4; ++mt) {
#pragma unroll
    for (int nt = 0; nt < 4; ++nt) {
#pragma unroll
      for (int i = 0; i < 8; ++i) {
        int m = mt * 16 + (lane >> 4) * 8 + i;
        int c = wave * 64 + nt * 16 + bN;
        sY[(size_t)m * YLD + c] = yacc[mt * 4 + nt][i];
      }
    }
  }
  __syncthreads();

  // writeout: out[tok] = / += comb * (y + proj_b)
  {
    const int r   = tid >> 3;
    const int tok = sTok[r];
    if (tok >= 0) {
      const float cw = sComb[r];
      const int c0 = (tid & 7) * 128;
      float* dst = out + (size_t)tok * C + c0;
      const float* bias = pb + (size_t)e * C + c0;
      for (int j = 0; j < 128; j += 4) {
        float4 yv = *(const float4*)&sY[(size_t)r * YLD + c0 + j];
        float4 bv = *(const float4*)(bias + j);
        float4 o;
        o.x = cw * (yv.x + bv.x);
        o.y = cw * (yv.y + bv.y);
        o.z = cw * (yv.z + bv.z);
        o.w = cw * (yv.w + bv.w);
        if (slot) {
          float4 prev = *(const float4*)(dst + j);
          o.x += prev.x; o.y += prev.y; o.z += prev.z; o.w += prev.w;
        }
        *(float4*)(dst + j) = o;
      }
    }
  }
}

// ---------------- host-side launch ----------------
extern "C" void kernel_launch(void* const* d_in, const int* in_sizes, int n_in,
                              void* d_out, int out_size, void* d_ws, size_t ws_size,
                              hipStream_t stream) {
  const float* x   = (const float*)d_in[0];
  const float* gw  = (const float*)d_in[1];
  const float* fcw = (const float*)d_in[2];
  const float* fcb = (const float*)d_in[3];
  const float* pw  = (const float*)d_in[4];
  const float* pb  = (const float*)d_in[5];
  float* out = (float*)d_out;

  // workspace carve
  float* comb   = (float*)d_ws;          // T*E
  float* probs  = comb + (size_t)T * E;  // T*E
  float* psum   = probs + (size_t)T * E; // E
  int*   counts = (int*)(psum + E);      // E*2
  int*   lists  = counts + E * 2;        // E*2*T
  char*  tail   = (char*)(lists + E * 2 * T);   // 16B-aligned by construction

  const size_t nX = (size_t)T * C;       // 4.19M elems
  const size_t nW = (size_t)E * F * C;   // 33.55M elems
  __bf16* xb   = (__bf16*)tail;
  __bf16* fcwb = xb + nX;
  __bf16* pwb  = fcwb + nW;
  const size_t ws_needed = (size_t)((char*)(pwb + nW) - (char*)d_ws);
  const bool preconv = ws_size >= ws_needed;   // fixed across calls -> deterministic

  hipMemsetAsync(counts, 0, E * 2 * sizeof(int), stream);

  if (preconv) {
    cvt_f32_to_bf16<<<(int)(nX / 2048), 256, 0, stream>>>(x, xb, (int)nX);
    cvt_f32_to_bf16<<<(int)(nW / 2048), 256, 0, stream>>>(fcw, fcwb, (int)nW);
    cvt_f32_to_bf16<<<(int)(nW / 2048), 256, 0, stream>>>(pw, pwb, (int)nW);
  }

  moe_router<<<T / 8, 256, 0, stream>>>(x, gw, comb, probs, counts, lists);
  moe_probs_sum<<<E, 256, 0, stream>>>(probs, psum);
  moe_aux<<<1, 32, 0, stream>>>(counts, psum, out + (size_t)T * C);

  dim3 grid(T / BM, E);
  if (preconv) {
    moe_expert_pass<true><<<grid, 512, SMEM_BYTES, stream>>>(
        x, xb, fcw, fcwb, fcb, pw, pwb, pb, comb, counts, lists, out, 0);
    moe_expert_pass<true><<<grid, 512, SMEM_BYTES, stream>>>(
        x, xb, fcw, fcwb, fcb, pw, pwb, pb, comb, counts, lists, out, 1);
  } else {
    moe_expert_pass<false><<<grid, 512, SMEM_BYTES, stream>>>(
        x, nullptr, fcw, nullptr, fcb, pw, nullptr, pb, comb, counts, lists, out, 0);
    moe_expert_pass<false><<<grid, 512, SMEM_BYTES, stream>>>(
        x, nullptr, fcw, nullptr, fcb, pw, nullptr, pb, comb, counts, lists, out, 1);
  }
}
